// HiddenToLogits_87101936763294
// MI455X (gfx1250) — matrix operand order repaired and checked by reading, not compiled, not run on hardware
//
#include <hip/hip_runtime.h>

// CDNA5 (gfx1250) wave32 WMMA types
typedef _Float16 v16h __attribute__((ext_vector_type(16)));
typedef float    v8f  __attribute__((ext_vector_type(8)));

#define BATCH   16384
#define HIDDEN  1024
#define NOUT    1880
#define MOVES   128

#define BM      128     // batch-rows per block
#define BN      128     // output-cols per block
#define BK      32      // K per stage == WMMA K
#define KSTEPS  (HIDDEN / BK)   // 32
#define LDH     36      // half stride for sA/sB rows (padded vs 32)
#define LDC     132     // float stride for sC rows (padded vs 128)

#define STAGE_BYTES ((BM + BN) * LDH * 2)   // 18432 B per pipeline stage (A+B)
#define SMEM_BYTES  (BM * LDC * 4)          // 67584 B; >= 2 stages (36864 B), reused as sC

__global__ __launch_bounds__(256)
void fused_gemm_gather_kernel(const float* __restrict__ hidden,
                              const float* __restrict__ weights,
                              const float* __restrict__ biases,
                              const int*   __restrict__ idx,
                              float*       __restrict__ out)
{
    __shared__ __align__(16) unsigned char smem[SMEM_BYTES];
    float* sC = (float*)smem;   // epilogue overlay

    const int tid  = threadIdx.x;
    const int lane = tid & 31;
    const int wave = tid >> 5;
    const int wm   = wave & 3;     // wave M index: 0..3  (32 rows each)
    const int wn   = wave >> 2;    // wave N index: 0..1  (64 cols each)

    const int n0 = blockIdx.x * BN;   // output-column tile base
    const int m0 = blockIdx.y * BM;   // batch-row tile base

    // ---- loader mapping (K-invariant): thread t loads rows rowBase + i*32, floats f4*4..+3 ----
    // Out-of-range weight rows are CLAMPED, not zeroed: GEMM columns n >= NOUT are never
    // consumed by the gather (idx < NOUT always), so any valid address is fine -> no
    // branches / exec-mask ops in the hot loader.
    const int rowBase = tid >> 3;     // 0..31
    const int f4      = tid & 7;      // 0..7
    const float* aPtrI[4];
    const float* bPtrI[4];
#pragma unroll
    for (int i = 0; i < 4; ++i) {
        const int ar = m0 + rowBase + i * 32;
        int       br = n0 + rowBase + i * 32;
        if (br > NOUT - 1) br = NOUT - 1;   // clamp (uniform, loop-invariant)
        aPtrI[i] = hidden  + (size_t)ar * HIDDEN + f4 * 4;
        bPtrI[i] = weights + (size_t)br * HIDDEN + f4 * 4;
    }

    float4 ga[4], gb[4];   // global prefetch registers

    auto loadTiles = [&](int k0) {
#pragma unroll
        for (int i = 0; i < 4; ++i) {
            ga[i] = *(const float4*)(aPtrI[i] + k0);
            gb[i] = *(const float4*)(bPtrI[i] + k0);
        }
    };
    auto storeTiles = [&](int p) {
        _Float16* sA = (_Float16*)(smem + p * STAGE_BYTES);
        _Float16* sB = sA + BM * LDH;
#pragma unroll
        for (int i = 0; i < 4; ++i) {
            union { _Float16 h[4]; unsigned long long u; } pa, pb;
            pa.h[0] = (_Float16)ga[i].x; pa.h[1] = (_Float16)ga[i].y;
            pa.h[2] = (_Float16)ga[i].z; pa.h[3] = (_Float16)ga[i].w;
            pb.h[0] = (_Float16)gb[i].x; pb.h[1] = (_Float16)gb[i].y;
            pb.h[2] = (_Float16)gb[i].z; pb.h[3] = (_Float16)gb[i].w;
            *(unsigned long long*)(sA + (rowBase + i * 32) * LDH + f4 * 4) = pa.u;
            *(unsigned long long*)(sB + (rowBase + i * 32) * LDH + f4 * 4) = pb.u;
        }
    };

    v8f acc[2][4];
#pragma unroll
    for (int mt = 0; mt < 2; ++mt)
#pragma unroll
        for (int nt = 0; nt < 4; ++nt)
            acc[mt][nt] = (v8f){0.f, 0.f, 0.f, 0.f, 0.f, 0.f, 0.f, 0.f};

    // ---- pipelined main loop: prefetch k+1 from global while computing k ----
    loadTiles(0);
    storeTiles(0);
    __syncthreads();

    for (int kt = 0; kt < KSTEPS; ++kt) {
        const int cur = kt & 1;
        if (kt + 1 < KSTEPS)
            loadTiles((kt + 1) * BK);   // in-flight during compute below

        const _Float16* sA = (const _Float16*)(smem + cur * STAGE_BYTES);
        const _Float16* sB = sA + BM * LDH;

        // ---- A fragments (16-bit A 16x32 layout, CDNA5 ISA 7.12.2) ----
        // lanes 0-15 : row M=lane,    VGPR v holds K = {(v&3)*2 + (v>>2)*16, +1}
        // lanes 16-31: row M=lane-16, same +8
        v16h afrag[2];
#pragma unroll
        for (int mt = 0; mt < 2; ++mt) {
            union { v16h h; unsigned int u[8]; } fa;
            const int arow = (wm * 32 + mt * 16 + (lane & 15)) * LDH;
            const int kofs = (lane < 16) ? 0 : 8;
#pragma unroll
            for (int vv = 0; vv < 8; ++vv) {
                const int kk = ((vv & 3) * 2) + ((vv >> 2) * 16) + kofs;
                fa.u[vv] = *(const unsigned int*)(sA + arow + kk);
            }
            afrag[mt] = fa.h;
        }
        // ---- B fragments (16-bit B 32x16: same K order as A; lanes 0-15 K={0..7,16..23}, lanes 16-31 +8) ----
        v16h bfrag[4];
#pragma unroll
        for (int nt = 0; nt < 4; ++nt) {
            union { v16h h; unsigned int u[8]; } fb;
            const int brow = (wn * 64 + nt * 16 + (lane & 15)) * LDH;
            const int kofs = (lane < 16) ? 0 : 8;
#pragma unroll
            for (int vv = 0; vv < 8; ++vv) {
                const int kk = ((vv & 3) * 2) + ((vv >> 2) * 16) + kofs;
                fb.u[vv] = *(const unsigned int*)(sB + brow + kk);
            }
            bfrag[nt] = fb.h;
        }

        // ---- 8 WMMAs: D = A x B + C, f32 accumulate ----
#pragma unroll
        for (int mt = 0; mt < 2; ++mt)
#pragma unroll
            for (int nt = 0; nt < 4; ++nt)
                acc[mt][nt] = __builtin_amdgcn_wmma_f32_16x16x32_f16(
                    /*neg_a=*/false, afrag[mt],
                    /*neg_b=*/false, bfrag[nt],
                    /*c_mod=*/(short)0, acc[mt][nt],
                    /*reuse_a=*/false, /*reuse_b=*/false);

        if (kt + 1 < KSTEPS) {
            storeTiles((kt + 1) & 1);   // other buffer; readers of it synced last iteration
            __syncthreads();
        }
    }

    // ---- epilogue: accumulators -> LDS (C/D layout: VGPR r -> M = r + 8*(lane>=16), N = lane&15) ----
    __syncthreads();   // everyone done reading stage buffers before overlaying sC
#pragma unroll
    for (int mt = 0; mt < 2; ++mt) {
#pragma unroll
        for (int nt = 0; nt < 4; ++nt) {
            const int rbase = wm * 32 + mt * 16 + ((lane >> 4) << 3);
            const int col   = wn * 64 + nt * 16 + (lane & 15);
#pragma unroll
            for (int r = 0; r < 8; ++r)
                sC[(rbase + r) * LDC + col] = acc[mt][nt][r];
        }
    }
    __syncthreads();

    // ---- fused gather + bias: scan this tile's 128 rows x 128 moves (int4 index loads) ----
    // Each (b,m) index lands in exactly one N-tile -> exactly one writer; full coverage.
#pragma unroll
    for (int i = 0; i < (BM * MOVES / 4) / 256; ++i) {   // 16 iterations of int4
        const int l4  = tid + i * 256;       // 0..4095 int4 slots
        const int row = l4 >> 5;             // 0..127 within M tile
        const int m4  = l4 & 31;             // group of 4 moves
        const int b   = m0 + row;
        const int4 nv = *(const int4*)(idx + (size_t)b * MOVES + m4 * 4);
#pragma unroll
        for (int j = 0; j < 4; ++j) {
            const int n = (j == 0) ? nv.x : (j == 1) ? nv.y : (j == 2) ? nv.z : nv.w;
            const int c = n - n0;
            if ((unsigned)c < (unsigned)BN) {
                // write-once, never re-read: non-temporal store keeps L2 for the
                // re-read working set (hidden f32 tiles, weights, idx)
                __builtin_nontemporal_store(sC[row * LDC + c] + biases[n],
                                            &out[(size_t)b * MOVES + m4 * 4 + j]);
            }
        }
    }
}

extern "C" void kernel_launch(void* const* d_in, const int* in_sizes, int n_in,
                              void* d_out, int out_size, void* d_ws, size_t ws_size,
                              hipStream_t stream) {
    const float* hidden  = (const float*)d_in[0];   // [16384, 1024] f32
    const float* weights = (const float*)d_in[1];   // [1880, 1024]  f32
    const float* biases  = (const float*)d_in[2];   // [1880]        f32
    const int*   idx     = (const int*)d_in[3];     // [16384, 128]  i32
    float*       out     = (float*)d_out;           // [16384, 128]  f32

    dim3 grid((NOUT + BN - 1) / BN, BATCH / BM);    // (15, 128)
    dim3 block(256);
    fused_gemm_gather_kernel<<<grid, block, 0, stream>>>(hidden, weights, biases, idx, out);
}
